// Mol2SpecEGNN_61770219651085
// MI455X (gfx1250) — compile-verified, matrix operand order checked by reading, not run on hardware
//
#include <hip/hip_runtime.h>
#include <hip/hip_bf16.h>

// ---------------------------------------------------------------------------
// Mol2Spec EGNN forward for MI455X (gfx1250, wave32, WMMA + TDM).
// GEMMs: bf16 x bf16 -> f32 via v_wmma_f32_16x16x32_bf16.
// Edge-MLP first GEMM fuses the h[row]/h[col] gather using the Tensor Data
// Mover in gather mode (16 row indices per descriptor -> LDS tile).
// ---------------------------------------------------------------------------

typedef __attribute__((ext_vector_type(16))) __bf16 v16bf;
typedef __attribute__((ext_vector_type(8)))  float  v8f;
typedef __attribute__((ext_vector_type(4)))  unsigned int u32x4;
typedef __attribute__((ext_vector_type(8)))  int i32x8;
typedef __attribute__((ext_vector_type(4)))  int i32x4;

union BFrag {
  v16bf v;
  unsigned int u32[8];
};

__device__ __forceinline__ unsigned short to_bf16(float f) {
  unsigned int u = __float_as_uint(f);
  unsigned int r = u + 0x7FFFu + ((u >> 16) & 1u);   // round-to-nearest-even
  return (unsigned short)(r >> 16);
}

__device__ __forceinline__ float bf_to_f32(unsigned short h) {
  return __uint_as_float(((unsigned int)h) << 16);
}

// Load one wave32 bf16 A/B fragment (16x32 tile, row-major source).
// ISA 7.12.2: lane L owns row M = L%16; half hf = L>>4:
//   elems 0..7  -> K = kb + hf*8 + (0..7)
//   elems 8..15 -> K = kb + 16 + hf*8 + (0..7)
__device__ __forceinline__ void load_frag(BFrag& f, const unsigned short* rowp,
                                          int kb, int hf) {
  const uint4* q0 = reinterpret_cast<const uint4*>(rowp + kb + hf * 8);
  const uint4* q1 = reinterpret_cast<const uint4*>(rowp + kb + 16 + hf * 8);
  uint4 a = *q0;
  uint4 b = *q1;
  f.u32[0] = a.x; f.u32[1] = a.y; f.u32[2] = a.z; f.u32[3] = a.w;
  f.u32[4] = b.x; f.u32[5] = b.y; f.u32[6] = b.z; f.u32[7] = b.w;
}

// ---------------------------------------------------------------------------
// Generic bf16 WMMA GEMM: out[M x Nc] = act(A[M x K] * W^T + bias) (+res)
//   W is N-major (pre-transposed). One wave = one 16x16 C tile.
// ---------------------------------------------------------------------------
__global__ __launch_bounds__(256) void k_gemm_bf16(
    const unsigned short* __restrict__ A, int lda,
    const unsigned short* __restrict__ W, int ldw,
    const float* __restrict__ bias,
    int M16, int N16, int K,
    float* __restrict__ outF, int ldf,
    const float* __restrict__ res, int ldr,
    unsigned short* __restrict__ outB, int ldb,
    int act) {
  int wid  = (int)((blockIdx.x * blockDim.x + threadIdx.x) >> 5);
  int lane = threadIdx.x & 31;
  if (wid >= M16 * N16) return;   // uniform per wave
  int tm = wid / N16;
  int tn = wid % N16;
  int mn = lane & 15;
  int hf = lane >> 4;

  const unsigned short* arow = A + (size_t)(tm * 16 + mn) * lda;
  const unsigned short* wrow = W + (size_t)(tn * 16 + mn) * ldw;

  v8f acc = {0.f, 0.f, 0.f, 0.f, 0.f, 0.f, 0.f, 0.f};
#pragma unroll 4
  for (int kb = 0; kb < K; kb += 32) {
    BFrag a, b;
    load_frag(a, arow, kb, hf);
    load_frag(b, wrow, kb, hf);
    acc = __builtin_amdgcn_wmma_f32_16x16x32_bf16(
        false, a.v, false, b.v, (short)0, acc, false, false);
  }

  float bn = bias ? bias[tn * 16 + mn] : 0.0f;
#pragma unroll
  for (int v = 0; v < 8; ++v) {
    int gr = tm * 16 + hf * 8 + v;
    int gc = tn * 16 + mn;
    float x = acc[v] + bn;
    if (act) x = x / (1.0f + __expf(-x)); // SiLU
    if (res)  x += res[(size_t)gr * ldr + gc];
    if (outF) outF[(size_t)gr * ldf + gc] = x;
    if (outB) outB[(size_t)gr * ldb + gc] = to_bf16(x);
  }
}

// ---------------------------------------------------------------------------
// Fused edge-MLP GEMM1 with TDM gather staging.
//   For a 16-edge tile: TDM gather-mode pulls h_bf[row[e]] and h_bf[col[e]]
//   (16 gathered 512-byte rows each) into LDS; 16 waves compute the full
//   16x256 output block; tail K-block (radial | edge_attr | pad) built in regs.
//   m1[e, :] = silu([h_r | h_c | radial | ea | 0pad] @ W1^T + b1)
// ---------------------------------------------------------------------------
__device__ __forceinline__ void tdm_gather_rows(unsigned lds_addr,
                                                const unsigned short* base,
                                                const unsigned int* idx /*16*/) {
  // D# group 0: count=1, gather_mode=1 (bit31), 16-bit indices (bit30=0),
  //             lds_addr in word1, global_addr[56:0] in words 2..3, type=2.
  unsigned long long ga = (unsigned long long)(size_t)base;
  u32x4 g0;
  g0[0] = 1u | (1u << 31);
  g0[1] = lds_addr;
  g0[2] = (unsigned)ga;
  g0[3] = ((unsigned)(ga >> 32) & 0x01FFFFFFu) | (2u << 30);
  // D# group 1: data_size=3 (8B units); row width 512B -> tile_dim0=64;
  //             tensor_dim0=64, tensor_dim1=N(rows); tile_dim1=16 indices;
  //             tensor_dim0_stride=64 (8B units).
  i32x8 g1;
  g1[0] = (int)(3u << 16);
  g1[1] = (int)(64u << 16);                // tensor_dim0 low16 at bits[31:16]
  g1[2] = (int)((20000u & 0xFFFFu) << 16); // tensor_dim1 low16 at bits[31:16]
  g1[3] = (int)(64u << 16);                // tile_dim0 at bits[31:16]
  g1[4] = 16;                              // tile_dim1 = #valid indices
  g1[5] = 64;                              // tensor_dim0_stride low32
  g1[6] = 0;
  g1[7] = 0;
  // D# groups 2/3: 16-bit row indices 0..7 / 8..15.
  i32x4 g2, g3;
  g2[0] = (int)(idx[0]  | (idx[1]  << 16));
  g2[1] = (int)(idx[2]  | (idx[3]  << 16));
  g2[2] = (int)(idx[4]  | (idx[5]  << 16));
  g2[3] = (int)(idx[6]  | (idx[7]  << 16));
  g3[0] = (int)(idx[8]  | (idx[9]  << 16));
  g3[1] = (int)(idx[10] | (idx[11] << 16));
  g3[2] = (int)(idx[12] | (idx[13] << 16));
  g3[3] = (int)(idx[14] | (idx[15] << 16));
  // 6-arg toolchain form: trailing int32x8 is the unused VADDR4 group (zeros).
  i32x8 g4 = {0, 0, 0, 0, 0, 0, 0, 0};
  __builtin_amdgcn_tensor_load_to_lds(g0, g1, g2, g3, g4, 0);
}

__global__ __launch_bounds__(512) void k_gemm_edge_tdm(
    const unsigned short* __restrict__ hbf,
    const int* __restrict__ row, const int* __restrict__ col,
    const float* __restrict__ radial, const float* __restrict__ ea,
    const unsigned short* __restrict__ W /*256 x 544 N-major*/,
    const float* __restrict__ bias,
    unsigned short* __restrict__ outB) {
  __shared__ __align__(16) unsigned short As[2][16 * 256];  // 2 x 8KB

  int tm    = blockIdx.x;
  int wave  = threadIdx.x >> 5;
  int lane  = threadIdx.x & 31;
  int mn    = lane & 15;
  int hf    = lane >> 4;
  int ebase = tm * 16;

  if (wave == 0) {
    unsigned ir[16], ic[16];
#pragma unroll
    for (int j = 0; j < 16; ++j) {
      ir[j] = (unsigned)row[ebase + j];
      ic[j] = (unsigned)col[ebase + j];
    }
    tdm_gather_rows((unsigned)(size_t)&As[0][0], hbf, ir);
    tdm_gather_rows((unsigned)(size_t)&As[1][0], hbf, ic);
    __builtin_amdgcn_s_wait_tensorcnt(0);
  }
  __syncthreads();

  int tn = wave;                                    // 16 waves -> tn 0..15
  const unsigned short* wrow = W + (size_t)(tn * 16 + mn) * 544;
  const unsigned short* ar   = &As[0][mn * 256];
  const unsigned short* ac   = &As[1][mn * 256];

  v8f acc = {0.f, 0.f, 0.f, 0.f, 0.f, 0.f, 0.f, 0.f};
#pragma unroll 4
  for (int kb = 0; kb < 256; kb += 32) {            // h[row] block
    BFrag a, b;
    load_frag(a, ar, kb, hf);
    load_frag(b, wrow, kb, hf);
    acc = __builtin_amdgcn_wmma_f32_16x16x32_bf16(
        false, a.v, false, b.v, (short)0, acc, false, false);
  }
#pragma unroll 4
  for (int kb = 0; kb < 256; kb += 32) {            // h[col] block
    BFrag a, b;
    load_frag(a, ac, kb, hf);
    load_frag(b, wrow, kb + 256, hf);
    acc = __builtin_amdgcn_wmma_f32_16x16x32_bf16(
        false, a.v, false, b.v, (short)0, acc, false, false);
  }
  {                                                 // tail K=512..543
    int e = ebase + mn;
    float rad = radial[e];
    unsigned short t16[16];
#pragma unroll
    for (int i = 0; i < 16; ++i) {
      int K = (i < 8) ? (512 + hf * 8 + i) : (528 + hf * 8 + (i - 8));
      float f = (K == 512) ? rad
              : ((K >= 513 && K <= 528) ? ea[(size_t)e * 16 + (K - 513)] : 0.0f);
      t16[i] = to_bf16(f);
    }
    BFrag a, b;
#pragma unroll
    for (int w = 0; w < 8; ++w)
      a.u32[w] = (unsigned)t16[2 * w] | ((unsigned)t16[2 * w + 1] << 16);
    load_frag(b, wrow, 512, hf);
    acc = __builtin_amdgcn_wmma_f32_16x16x32_bf16(
        false, a.v, false, b.v, (short)0, acc, false, false);
  }

  float bn = bias[tn * 16 + mn];
#pragma unroll
  for (int v = 0; v < 8; ++v) {
    int gr = tm * 16 + hf * 8 + v;
    int gc = tn * 16 + mn;
    float x = acc[v] + bn;
    x = x / (1.0f + __expf(-x));                    // SiLU
    outB[(size_t)gr * 256 + gc] = to_bf16(x);
  }
}

// --------------------------- utility kernels -------------------------------

__global__ void k_zero32(unsigned int* __restrict__ p, int n) {
  int i = blockIdx.x * 256 + threadIdx.x;
  if (i < n) p[i] = 0u;
}

__global__ void k_copy_f32(const float* __restrict__ s, float* __restrict__ d, int n) {
  int i = blockIdx.x * 256 + threadIdx.x;
  if (i < n) d[i] = s[i];
}

__global__ void k_f2bf(const float* __restrict__ s, unsigned short* __restrict__ d, int n) {
  int i = blockIdx.x * 256 + threadIdx.x;
  if (i < n) d[i] = to_bf16(s[i]);
}

// Wt[n*Kpad + k] = (k < K) ? W[k*Nc + n] : 0   (f32 KxNc -> bf16 Nc x Kpad)
__global__ void k_transpose_w(const float* __restrict__ W, unsigned short* __restrict__ Wt,
                              int K, int Nc, int Kpad) {
  int i = blockIdx.x * 256 + threadIdx.x;
  if (i >= Nc * Kpad) return;
  int nn = i / Kpad, kk = i % Kpad;
  float v = (kk < K) ? W[(size_t)kk * Nc + nn] : 0.0f;
  Wt[i] = to_bf16(v);
}

__global__ void k_count(const int* __restrict__ row, float* __restrict__ cnt, int E) {
  int e = blockIdx.x * 256 + threadIdx.x;
  if (e < E) atomicAdd(&cnt[row[e]], 1.0f);
}

__global__ void k_edgegeo(const float* __restrict__ coord, const int* __restrict__ row,
                          const int* __restrict__ col, float* __restrict__ diff,
                          float* __restrict__ radial, int E) {
  int e = blockIdx.x * 256 + threadIdx.x;
  if (e >= E) return;
  int r = row[e], c = col[e];
  float dx = coord[r * 3 + 0] - coord[c * 3 + 0];
  float dy = coord[r * 3 + 1] - coord[c * 3 + 1];
  float dz = coord[r * 3 + 2] - coord[c * 3 + 2];
  diff[(size_t)e * 3 + 0] = dx;
  diff[(size_t)e * 3 + 1] = dy;
  diff[(size_t)e * 3 + 2] = dz;
  radial[e] = dx * dx + dy * dy + dz * dz;
}

// c[e] = sum_k t[e,k] * w[k]   (8 lanes per edge, shuffle reduce)
__global__ void k_cdot(const unsigned short* __restrict__ tb, const float* __restrict__ w,
                       float* __restrict__ cvec, int E) {
  size_t t = (size_t)blockIdx.x * 256 + threadIdx.x;
  if (t >= (size_t)E * 8) return;
  int e = (int)(t >> 3);
  int j = (int)(t & 7);
  const unsigned short* p = tb + (size_t)e * 256 + j * 32;
  float s = 0.0f;
#pragma unroll
  for (int k = 0; k < 32; ++k) s += bf_to_f32(p[k]) * w[j * 32 + k];
  s += __shfl_xor(s, 1, 8);
  s += __shfl_xor(s, 2, 8);
  s += __shfl_xor(s, 4, 8);
  if (j == 0) cvec[e] = s;
}

__global__ void k_scatter_coord(const float* __restrict__ diff, const float* __restrict__ cvec,
                                const int* __restrict__ row, float* __restrict__ cagg, int E) {
  int e = blockIdx.x * 256 + threadIdx.x;
  if (e >= E) return;
  float cc = cvec[e];
  int r = row[e];
  atomicAdd(&cagg[r * 3 + 0], diff[(size_t)e * 3 + 0] * cc);
  atomicAdd(&cagg[r * 3 + 1], diff[(size_t)e * 3 + 1] * cc);
  atomicAdd(&cagg[r * 3 + 2], diff[(size_t)e * 3 + 2] * cc);
}

__global__ void k_coord_apply(float* __restrict__ coord, const float* __restrict__ cagg,
                              const float* __restrict__ cnt, int N) {
  int t = blockIdx.x * 256 + threadIdx.x;
  if (t >= N * 3) return;
  coord[t] += cagg[t] / fmaxf(cnt[t / 3], 1.0f);
}

__global__ void k_scatter_add(const unsigned short* __restrict__ m, const int* __restrict__ row,
                              float* __restrict__ agg, int E) {
  size_t t = (size_t)blockIdx.x * 256 + threadIdx.x;
  if (t >= (size_t)E * 256) return;
  int e = (int)(t >> 8);
  int c = (int)(t & 255);
  atomicAdd(&agg[(size_t)row[e] * 256 + c], bf_to_f32(m[t]));
}

__global__ void k_build_nodecat(const unsigned short* __restrict__ hbf,
                                const float* __restrict__ agg,
                                unsigned short* __restrict__ cat, int N) {
  size_t t = (size_t)blockIdx.x * 256 + threadIdx.x;
  if (t >= (size_t)N * 512) return;
  int n = (int)(t >> 9);
  int c = (int)(t & 511);
  cat[t] = (c < 256) ? hbf[(size_t)n * 256 + c] : to_bf16(agg[(size_t)n * 256 + (c - 256)]);
}

__global__ void k_graph_start(const int* __restrict__ batch, int* __restrict__ start,
                              int N, int B) {
  int b = blockIdx.x * 256 + threadIdx.x;
  if (b > B) return;
  if (b == B) { start[B] = N; return; }
  int lo = 0, hi = N;
  while (lo < hi) {
    int mid = (lo + hi) >> 1;
    if (batch[mid] < b) lo = mid + 1; else hi = mid;
  }
  start[b] = lo;
}

__global__ void k_pool(const float* __restrict__ hout, const int* __restrict__ start,
                       float* __restrict__ pooled, int B) {
  int t = blockIdx.x * 256 + threadIdx.x;
  if (t >= B * 256) return;
  int b = t >> 8, j = t & 255;
  int s = start[b], e = start[b + 1];
  float m = hout[(size_t)s * 256 + j];
  for (int i = s + 1; i < e; ++i) {
    if (i + 16 < e) __builtin_prefetch(&hout[(size_t)(i + 16) * 256 + j], 0, 0);
    m = fmaxf(m, hout[(size_t)i * 256 + j]);
  }
  pooled[t] = m;
}

// z = [pooled(256) | frag(8) | adduct(8)]; zb padded to 288 with zeros
__global__ void k_build_z(const float* __restrict__ pooled, const float* __restrict__ frag,
                          const float* __restrict__ adduct, float* __restrict__ zf,
                          unsigned short* __restrict__ zb, int B) {
  int t = blockIdx.x * 256 + threadIdx.x;
  if (t >= B * 288) return;
  int b = t / 288, c = t % 288;
  float f;
  if (c < 256)      f = pooled[b * 256 + c];
  else if (c < 264) f = frag[b * 8 + (c - 256)];
  else if (c < 272) f = adduct[b * 8 + (c - 264)];
  else              f = 0.0f;
  zb[t] = to_bf16(f);
  if (c < 272) zf[b * 272 + c] = f;
}

// ---------------------------------------------------------------------------

#define CDIV(a, b) (((a) + (b)-1) / (b))

extern "C" void kernel_launch(void* const* d_in, const int* in_sizes, int n_in,
                              void* d_out, int out_size, void* d_ws, size_t ws_size,
                              hipStream_t stream) {
  const int N = 20000, E = 320000, B = 128, H = 256, L = 4;

  const float* x          = (const float*)d_in[0];
  const float* pos        = (const float*)d_in[1];
  const float* edge_attr  = (const float*)d_in[2];
  const float* frag       = (const float*)d_in[3];
  const float* adduct     = (const float*)d_in[4];
  const int*   eidx       = (const int*)d_in[5];
  const int*   batch      = (const int*)d_in[6];
  const float* emb_in_w   = (const float*)d_in[7];
  const float* emb_in_b   = (const float*)d_in[8];
  const float* edge_w1    = (const float*)d_in[9];
  const float* edge_b1    = (const float*)d_in[10];
  const float* edge_w2    = (const float*)d_in[11];
  const float* edge_b2    = (const float*)d_in[12];
  const float* node_w1    = (const float*)d_in[13];
  const float* node_b1    = (const float*)d_in[14];
  const float* node_w2    = (const float*)d_in[15];
  const float* node_b2    = (const float*)d_in[16];
  const float* coord_w1   = (const float*)d_in[17];
  const float* coord_b1   = (const float*)d_in[18];
  const float* coord_w2   = (const float*)d_in[19];
  const float* emb_out_w  = (const float*)d_in[20];
  const float* emb_out_b  = (const float*)d_in[21];
  const float* res_w1     = (const float*)d_in[22];
  const float* res_b1     = (const float*)d_in[23];
  const float* res_w2     = (const float*)d_in[24];
  const float* res_b2     = (const float*)d_in[25];
  const float* head_w     = (const float*)d_in[26];
  const float* head_b     = (const float*)d_in[27];
  const int* row = eidx;
  const int* col = eidx + E;
  float* out = (float*)d_out;

  // ---- workspace layout (bump allocator, 256B aligned) ----
  char* ws = (char*)d_ws;
  size_t off = 0;
  auto alloc = [&](size_t bytes) -> void* {
    void* p = ws + off;
    off = (off + bytes + 255) & ~(size_t)255;
    return p;
  };
  unsigned short* x_bf    = (unsigned short*)alloc((size_t)N * 128 * 2);
  float*          h       = (float*)alloc((size_t)N * H * 4);
  unsigned short* h_bf    = (unsigned short*)alloc((size_t)N * H * 2);
  float*          agg     = (float*)alloc((size_t)N * H * 4);   // also reused as hout
  unsigned short* nodecat = (unsigned short*)alloc((size_t)N * 512 * 2);
  unsigned short* nodetmp = (unsigned short*)alloc((size_t)N * H * 2);
  float*          coord   = (float*)alloc((size_t)N * 3 * 4);
  float*          cagg    = (float*)alloc((size_t)N * 3 * 4);
  float*          cnt     = (float*)alloc((size_t)N * 4);
  float*          diff    = (float*)alloc((size_t)E * 3 * 4);
  float*          radial  = (float*)alloc((size_t)E * 4);
  float*          cvec    = (float*)alloc((size_t)E * 4);
  unsigned short* m1      = (unsigned short*)alloc((size_t)E * H * 2);
  unsigned short* m2      = (unsigned short*)alloc((size_t)E * H * 2);
  unsigned short* tbuf    = m1;          // alias: m1 dead after GEMM2
  int*            start   = (int*)alloc((size_t)(B + 1) * 4);
  float*          pooled  = (float*)alloc((size_t)B * H * 4);
  float*          zf      = (float*)alloc((size_t)B * 272 * 4);
  unsigned short* zb      = (unsigned short*)alloc((size_t)B * 288 * 2);
  unsigned short* r1      = (unsigned short*)alloc((size_t)B * H * 2);
  float*          z2f     = (float*)alloc((size_t)B * 272 * 4);
  unsigned short* z2b     = (unsigned short*)alloc((size_t)B * 288 * 2);
  // bf16 N-major (transposed) weights
  unsigned short* emb_in_wt  = (unsigned short*)alloc((size_t)256 * 128 * 2);
  unsigned short* edge_w1t   = (unsigned short*)alloc((size_t)L * 256 * 544 * 2);
  unsigned short* edge_w2t   = (unsigned short*)alloc((size_t)L * 256 * 256 * 2);
  unsigned short* coord_w1t  = (unsigned short*)alloc((size_t)L * 256 * 256 * 2);
  unsigned short* node_w1t   = (unsigned short*)alloc((size_t)L * 256 * 512 * 2);
  unsigned short* node_w2t   = (unsigned short*)alloc((size_t)L * 256 * 256 * 2);
  unsigned short* emb_out_wt = (unsigned short*)alloc((size_t)256 * 256 * 2);
  unsigned short* res_w1t    = (unsigned short*)alloc((size_t)256 * 288 * 2);
  unsigned short* res_w2t    = (unsigned short*)alloc((size_t)272 * 256 * 2);
  unsigned short* head_wt    = (unsigned short*)alloc((size_t)2000 * 288 * 2);

  dim3 blk(256);
  auto G1 = [](long long n) { return dim3((unsigned)CDIV(n, 256)); };
  auto GW = [](int tiles) { return dim3((unsigned)CDIV(tiles, 8)); };  // 8 waves/block

  // ---- weight conversion / transposition ----
  k_transpose_w<<<G1(256 * 128), blk, 0, stream>>>(emb_in_w, emb_in_wt, 128, 256, 128);
  for (int l = 0; l < L; ++l) {
    k_transpose_w<<<G1(256 * 544), blk, 0, stream>>>(edge_w1 + (size_t)l * 529 * 256,
                                                     edge_w1t + (size_t)l * 256 * 544, 529, 256, 544);
    k_transpose_w<<<G1(256 * 256), blk, 0, stream>>>(edge_w2 + (size_t)l * 256 * 256,
                                                     edge_w2t + (size_t)l * 256 * 256, 256, 256, 256);
    k_transpose_w<<<G1(256 * 256), blk, 0, stream>>>(coord_w1 + (size_t)l * 256 * 256,
                                                     coord_w1t + (size_t)l * 256 * 256, 256, 256, 256);
    k_transpose_w<<<G1(256 * 512), blk, 0, stream>>>(node_w1 + (size_t)l * 512 * 256,
                                                     node_w1t + (size_t)l * 256 * 512, 512, 256, 512);
    k_transpose_w<<<G1(256 * 256), blk, 0, stream>>>(node_w2 + (size_t)l * 256 * 256,
                                                     node_w2t + (size_t)l * 256 * 256, 256, 256, 256);
  }
  k_transpose_w<<<G1(256 * 256), blk, 0, stream>>>(emb_out_w, emb_out_wt, 256, 256, 256);
  k_transpose_w<<<G1(256 * 288), blk, 0, stream>>>(res_w1, res_w1t, 272, 256, 288);
  k_transpose_w<<<G1(272 * 256), blk, 0, stream>>>(res_w2, res_w2t, 256, 272, 256);
  k_transpose_w<<<G1(2000 * 288), blk, 0, stream>>>(head_w, head_wt, 272, 2000, 288);

  // ---- setup: coord, cnt, x_bf ----
  k_copy_f32<<<G1(N * 3), blk, 0, stream>>>(pos, coord, N * 3);
  k_zero32<<<G1(N), blk, 0, stream>>>((unsigned int*)cnt, N);
  k_count<<<G1(E), blk, 0, stream>>>(row, cnt, E);
  k_f2bf<<<G1((long long)N * 128), blk, 0, stream>>>(x, x_bf, N * 128);

  // ---- h = x @ emb_in_w + b ----
  k_gemm_bf16<<<GW(1250 * 16), blk, 0, stream>>>(x_bf, 128, emb_in_wt, 128, emb_in_b,
                                                 1250, 16, 128, h, 256, nullptr, 0,
                                                 h_bf, 256, 0);

  // ---- EGNN layers ----
  for (int l = 0; l < L; ++l) {
    k_edgegeo<<<G1(E), blk, 0, stream>>>(coord, row, col, diff, radial, E);
    // m = silu([h_r|h_c|radial|ea] @ W1 + b1)  -- fused gather via TDM
    k_gemm_edge_tdm<<<dim3(20000), dim3(512), 0, stream>>>(
        h_bf, row, col, radial, edge_attr,
        edge_w1t + (size_t)l * 256 * 544, edge_b1 + l * 256, m1);
    // m = silu(m @ W2 + b2)
    k_gemm_bf16<<<GW(20000 * 16), blk, 0, stream>>>(m1, 256, edge_w2t + (size_t)l * 256 * 256,
                                                    256, edge_b2 + l * 256, 20000, 16, 256,
                                                    nullptr, 0, nullptr, 0, m2, 256, 1);
    // t = silu(m @ coord_w1 + b)   (tbuf aliases m1)
    k_gemm_bf16<<<GW(20000 * 16), blk, 0, stream>>>(m2, 256, coord_w1t + (size_t)l * 256 * 256,
                                                    256, coord_b1 + l * 256, 20000, 16, 256,
                                                    nullptr, 0, nullptr, 0, tbuf, 256, 1);
    // c = t @ coord_w2 ; coord += segsum(diff*c)/cnt
    k_cdot<<<G1((long long)E * 8), blk, 0, stream>>>(tbuf, coord_w2 + l * 256, cvec, E);
    k_zero32<<<G1(N * 3), blk, 0, stream>>>((unsigned int*)cagg, N * 3);
    k_scatter_coord<<<G1(E), blk, 0, stream>>>(diff, cvec, row, cagg, E);
    k_coord_apply<<<G1(N * 3), blk, 0, stream>>>(coord, cagg, cnt, N);
    // agg = segsum(m)
    k_zero32<<<G1((long long)N * 256), blk, 0, stream>>>((unsigned int*)agg, N * 256);
    k_scatter_add<<<G1((long long)E * 256), blk, 0, stream>>>(m2, row, agg, E);
    // node MLP + residual
    k_build_nodecat<<<G1((long long)N * 512), blk, 0, stream>>>(h_bf, agg, nodecat, N);
    k_gemm_bf16<<<GW(1250 * 16), blk, 0, stream>>>(nodecat, 512, node_w1t + (size_t)l * 256 * 512,
                                                   512, node_b1 + l * 256, 1250, 16, 512,
                                                   nullptr, 0, nullptr, 0, nodetmp, 256, 1);
    k_gemm_bf16<<<GW(1250 * 16), blk, 0, stream>>>(nodetmp, 256, node_w2t + (size_t)l * 256 * 256,
                                                   256, node_b2 + l * 256, 1250, 16, 256,
                                                   h, 256, h, 256, h_bf, 256, 0);
  }

  // ---- output embed, pool, res-block, head ----
  k_gemm_bf16<<<GW(1250 * 16), blk, 0, stream>>>(h_bf, 256, emb_out_wt, 256, emb_out_b,
                                                 1250, 16, 256, agg /*hout*/, 256,
                                                 nullptr, 0, nullptr, 0, 0);
  k_graph_start<<<G1(B + 1), blk, 0, stream>>>(batch, start, N, B);
  k_pool<<<G1(B * 256), blk, 0, stream>>>(agg, start, pooled, B);
  k_build_z<<<G1(B * 288), blk, 0, stream>>>(pooled, frag, adduct, zf, zb, B);
  // r1 = silu(z @ res_w1 + b)
  k_gemm_bf16<<<GW(8 * 16), blk, 0, stream>>>(zb, 288, res_w1t, 288, res_b1,
                                              8, 16, 288, nullptr, 0, nullptr, 0, r1, 256, 1);
  // z2 = z + (r1 @ res_w2 + b2)
  k_zero32<<<G1(B * 288 / 2), blk, 0, stream>>>((unsigned int*)z2b, B * 288 / 2);
  k_gemm_bf16<<<GW(8 * 17), blk, 0, stream>>>(r1, 256, res_w2t, 256, res_b2,
                                              8, 17, 256, z2f, 272, zf, 272, z2b, 288, 0);
  // out = z2 @ head_w + head_b   -> [128 x 2000]
  k_gemm_bf16<<<GW(8 * 125), blk, 0, stream>>>(z2b, 288, head_wt, 288, head_b,
                                               8, 125, 288, out, 2000, nullptr, 0,
                                               nullptr, 0, 0);
  (void)in_sizes; (void)n_in; (void)out_size; (void)ws_size;
}
